// LRULayer_14087492731469
// MI455X (gfx1250) — matrix operand order repaired and checked by reading, not compiled, not run on hardware
//
#include <hip/hip_runtime.h>
#include <hip/hip_bf16.h>

// ---------------------------------------------------------------------------
// LRU layer for MI455X (gfx1250, wave32, WMMA + Tensor Data Mover).
//   Wbf  = bf16(W_in), bf16(W_out)                  (one-time, 4 MB)
//   gu   = bf16( gamma * (x @ W_in^T + b_in) )      bf16-WMMA GEMM, TDM-fed W
//   y_t  = lam * y_{t-1} + gu_t  (fp32 carry)       in-place bf16 scan
//   out  = y @ W_out^T + b_out + x                  bf16-WMMA GEMM, TDM-fed A+W
// ---------------------------------------------------------------------------

#define B_DIM 32
#define L_DIM 2048
#define D_DIM 1024
#define M_TOT (B_DIM * L_DIM)   // 65536 rows

typedef __bf16 bf16_t;
typedef bf16_t v16bf __attribute__((ext_vector_type(16)));
typedef float  v8f   __attribute__((ext_vector_type(8)));
typedef unsigned int v8u  __attribute__((ext_vector_type(8)));
typedef unsigned int v4u  __attribute__((ext_vector_type(4)));
typedef int          v8i  __attribute__((ext_vector_type(8)));
typedef int          v4i  __attribute__((ext_vector_type(4)));

__device__ __forceinline__ unsigned int f2bf_rne(float f) {
    unsigned int u = __builtin_bit_cast(unsigned int, f);
    u += 0x7FFFu + ((u >> 16) & 1u);          // round to nearest even
    return u >> 16;
}
__device__ __forceinline__ unsigned int pack_bf2(float lo, float hi) {
#if __has_builtin(__builtin_amdgcn_cvt_pk_bf16_f32)
    return __builtin_bit_cast(unsigned int,
                              __builtin_amdgcn_cvt_pk_bf16_f32(lo, hi));
#else
    return f2bf_rne(lo) | (f2bf_rne(hi) << 16);
#endif
}
__device__ __forceinline__ float bf2f(unsigned int b16) {
    return __builtin_bit_cast(float, b16 << 16);
}

#define BM 128
#define BN 64
#define BK 32
#define LDA_PAD 34   // ushort row stride: 32 elems + 1 DWORD pad (TDM pad_amount=1)
#define NT (D_DIM / BK)

// ---------------------------------------------------------------------------
// Tensor Data Mover: 2D bf16 tile (tile_d1 rows x tile_d0 elems) -> LDS,
// LDS rows padded by 1 DWORD every 16 DWORDs (64B rows -> 68B stride).
// D# layout per CDNA5 ISA sec. 8.3/8.4; groups 2/3 zero (<=2D tensor).
// ---------------------------------------------------------------------------
__device__ __forceinline__ void tdm_load_2d_bf16(
    unsigned lds_byte_addr, const void* gtile,
    unsigned tensor_d0, unsigned tensor_d1, unsigned d0_stride,
    unsigned tile_d0, unsigned tile_d1)
{
    unsigned long long ga = (unsigned long long)(uintptr_t)gtile;
    v4u g0;
    g0[0] = 1u;                                                  // count=1 (user)
    g0[1] = lds_byte_addr;                                       // lds_addr
    g0[2] = (unsigned)(ga & 0xFFFFFFFFu);                        // global_addr lo
    g0[3] = (unsigned)((ga >> 32) & 0x01FFFFFFu) | (2u << 30);   // addr hi | type=2
    v8i g1;
    g1[0] = (int)((1u << 16) | (1u << 20) | (3u << 22));         // data_size=2B, pad_en, pad_int=16dw, pad_amt=1dw
    g1[1] = (int)((tensor_d0 & 0xFFFFu) << 16);                  // tensor_dim0[15:0]
    g1[2] = (int)(((tensor_d0 >> 16) & 0xFFFFu) | ((tensor_d1 & 0xFFFFu) << 16));
    g1[3] = (int)(((tensor_d1 >> 16) & 0xFFFFu) | ((tile_d0 & 0xFFFFu) << 16));
    g1[4] = (int)(tile_d1 & 0xFFFFu);                            // tile_dim1 | tile_dim2=0
    g1[5] = (int)d0_stride;                                      // dim0_stride[31:0]
    g1[6] = 0;
    g1[7] = 0;
    v4i g2 = {0, 0, 0, 0};
    v4i g3 = {0, 0, 0, 0};
#if __clang_major__ >= 23
    v8i g4 = {0, 0, 0, 0, 0, 0, 0, 0};
    __builtin_amdgcn_tensor_load_to_lds(g0, g1, g2, g3, g4, 0);
#else
    __builtin_amdgcn_tensor_load_to_lds(g0, g1, g2, g3, 0);
#endif
}

// ---- fragment gathers (ISA 7.12.2 layouts), LDS row stride = LDA_PAD ushorts
__device__ __forceinline__ v16bf frag_a(const unsigned short* Arow, int half) {
    const int ah = half * 8;
    v8u raw;
#pragma unroll
    for (int v = 0; v < 8; ++v) {
        int k = ((v < 4) ? (2 * v) : (8 + 2 * v)) + ah;  // 0,2,4,6,16,18,20,22 (+8)
        raw[v] = *(const unsigned int*)(Arow + k);
    }
    return __builtin_bit_cast(v16bf, raw);
}
__device__ __forceinline__ v16bf frag_b(const unsigned short* Brow, int half) {
    const int bh = half * 8;
    v8u raw;
#pragma unroll
    for (int v = 0; v < 8; ++v)
        raw[v] = *(const unsigned int*)(Brow + bh + ((v < 4) ? (2 * v) : (8 + 2 * v)));
    return __builtin_bit_cast(v16bf, raw);
}

// ---------------------------------------------------------------------------
// fp32 -> bf16 weight conversion (one-time, 1024x1024)
// ---------------------------------------------------------------------------
__global__ __launch_bounds__(256)
void w_to_bf16(const float* __restrict__ W, unsigned int* __restrict__ Wb) {
    const int i = blockIdx.x * blockDim.x + threadIdx.x;     // 0 .. 1024*1024/2-1
    const float2 v = *(const float2*)(W + 2 * (size_t)i);
    Wb[i] = pack_bf2(v.x, v.y);
}

// ---------------------------------------------------------------------------
// GEMM1: gu_bf16 = gamma * (x @ W_in^T + b_in).
// A (x, fp32) converted inline into LDS; W tile bf16 via TDM (1-deep pipeline).
// ---------------------------------------------------------------------------
__global__ __launch_bounds__(256)
void lru_gemm1(const float* __restrict__ A,            // x [M_TOT, D]
               const unsigned short* __restrict__ Wb,  // W_in bf16 [D, D]
               const float* __restrict__ bias,         // b_in
               const float* __restrict__ gamma,
               unsigned short* __restrict__ Out)       // gu bf16 [M_TOT, D]
{
    __shared__ unsigned short As[BM * LDA_PAD];
    __shared__ unsigned short Bs[2][BN * LDA_PAD];

    const int tid = threadIdx.x, wv = tid >> 5, lane = tid & 31;
    const int half = lane >> 4, lan16 = lane & 15;
    const int M0 = blockIdx.x * BM, N0 = blockIdx.y * BN;

    v8f acc[4];
#pragma unroll
    for (int i = 0; i < 4; ++i) acc[i] = (v8f){0.f,0.f,0.f,0.f,0.f,0.f,0.f,0.f};

    if (wv == 0)   // prime TDM: W tile for kt=0
        tdm_load_2d_bf16((unsigned)(uintptr_t)&Bs[0][0], Wb + (size_t)N0 * D_DIM,
                         D_DIM, D_DIM, D_DIM, BK, BN);

    for (int kt = 0; kt < NT; ++kt) {
        const int k0 = kt * BK, buf = kt & 1;
        // ---- A tile: 128x32 fp32 -> bf16 into LDS ----
#pragma unroll
        for (int i = 0; i < 4; ++i) {
            int idx = tid + i * 256, row = idx >> 3, c4 = (idx & 7) * 4;
            const float4 v = *(const float4*)(A + (size_t)(M0 + row) * D_DIM + k0 + c4);
            unsigned int* dst = (unsigned int*)&As[row * LDA_PAD + c4];
            dst[0] = pack_bf2(v.x, v.y);
            dst[1] = pack_bf2(v.z, v.w);
        }
        if (kt + 1 < NT) {
#pragma unroll
            for (int i = 0; i < 4; ++i) {
                int idx = tid + i * 256;
                __builtin_prefetch(A + (size_t)(M0 + (idx >> 3)) * D_DIM + (k0 + BK) + (idx & 7) * 4, 0, 1);
            }
        }
        if (wv == 0) {
            if (kt + 1 < NT) {   // issue next W tile into other buffer, keep 1 in flight
                tdm_load_2d_bf16((unsigned)(uintptr_t)&Bs[buf ^ 1][0],
                                 Wb + (size_t)N0 * D_DIM + (k0 + BK),
                                 D_DIM, D_DIM, D_DIM, BK, BN);
                __builtin_amdgcn_s_wait_tensorcnt(1);
            } else {
                __builtin_amdgcn_s_wait_tensorcnt(0);
            }
        }
        __syncthreads();

        const v16bf afrag = frag_a(&As[(wv * 16 + lan16) * LDA_PAD], half);
#pragma unroll
        for (int nb = 0; nb < 4; ++nb) {
            const v16bf bfrag = frag_b(&Bs[buf][(nb * 16 + lan16) * LDA_PAD], half);
            acc[nb] = __builtin_amdgcn_wmma_f32_16x16x32_bf16(
                false, afrag, false, bfrag, (short)0, acc[nb], false, false);
        }
        __syncthreads();
    }
    // epilogue: D layout -> m = r + 8*(lane>>4), n = lane&15
#pragma unroll
    for (int nb = 0; nb < 4; ++nb) {
        const int e = N0 + nb * 16 + lan16;
        const float bv = bias[e], g = gamma[e];
#pragma unroll
        for (int r = 0; r < 8; ++r) {
            const size_t m = (size_t)(M0 + wv * 16 + r + 8 * half);
            Out[m * D_DIM + e] = (unsigned short)(pack_bf2(g * (acc[nb][r] + bv), 0.f) & 0xFFFFu);
        }
    }
}

// ---------------------------------------------------------------------------
// GEMM2: out = y @ W_out^T + b_out + x.  A (y bf16) and W bf16 both TDM-fed,
// double-buffered: issue tile k+1, s_wait_tensorcnt(2), compute tile k.
// ---------------------------------------------------------------------------
__global__ __launch_bounds__(256)
void lru_gemm2(const unsigned short* __restrict__ Yb,  // y bf16 [M_TOT, D]
               const unsigned short* __restrict__ Wb,  // W_out bf16 [D, D]
               const float* __restrict__ bias,         // b_out
               const float* __restrict__ Xres,         // x (residual)
               float* __restrict__ Out)
{
    __shared__ unsigned short As[2][BM * LDA_PAD];
    __shared__ unsigned short Bs[2][BN * LDA_PAD];

    const int tid = threadIdx.x, wv = tid >> 5, lane = tid & 31;
    const int half = lane >> 4, lan16 = lane & 15;
    const int M0 = blockIdx.x * BM, N0 = blockIdx.y * BN;

    v8f acc[4];
#pragma unroll
    for (int i = 0; i < 4; ++i) acc[i] = (v8f){0.f,0.f,0.f,0.f,0.f,0.f,0.f,0.f};

    if (wv == 0) {  // prime tile 0
        tdm_load_2d_bf16((unsigned)(uintptr_t)&As[0][0], Yb + (size_t)M0 * D_DIM,
                         D_DIM, M_TOT, D_DIM, BK, BM);
        tdm_load_2d_bf16((unsigned)(uintptr_t)&Bs[0][0], Wb + (size_t)N0 * D_DIM,
                         D_DIM, D_DIM, D_DIM, BK, BN);
    }

    for (int kt = 0; kt < NT; ++kt) {
        const int k0 = kt * BK, buf = kt & 1;
        if (wv == 0) {
            if (kt + 1 < NT) {   // buf^1 free: all waves passed end-of-iter barrier kt-1
                tdm_load_2d_bf16((unsigned)(uintptr_t)&As[buf ^ 1][0],
                                 Yb + (size_t)M0 * D_DIM + (k0 + BK),
                                 D_DIM, M_TOT, D_DIM, BK, BM);
                tdm_load_2d_bf16((unsigned)(uintptr_t)&Bs[buf ^ 1][0],
                                 Wb + (size_t)N0 * D_DIM + (k0 + BK),
                                 D_DIM, D_DIM, D_DIM, BK, BN);
                __builtin_amdgcn_s_wait_tensorcnt(2);   // tile kt done; kt+1 may fly
            } else {
                __builtin_amdgcn_s_wait_tensorcnt(0);   // drain for the last tile
            }
        }
        __syncthreads();

        const v16bf afrag = frag_a(&As[buf][(wv * 16 + lan16) * LDA_PAD], half);
#pragma unroll
        for (int nb = 0; nb < 4; ++nb) {
            const v16bf bfrag = frag_b(&Bs[buf][(nb * 16 + lan16) * LDA_PAD], half);
            acc[nb] = __builtin_amdgcn_wmma_f32_16x16x32_bf16(
                false, afrag, false, bfrag, (short)0, acc[nb], false, false);
        }
        __syncthreads();
    }
#pragma unroll
    for (int nb = 0; nb < 4; ++nb) {
        const int e = N0 + nb * 16 + lan16;
        const float bv = bias[e];
#pragma unroll
        for (int r = 0; r < 8; ++r) {
            const size_t m = (size_t)(M0 + wv * 16 + r + 8 * half);
            const size_t off = m * D_DIM + e;
            Out[off] = acc[nb][r] + bv + Xres[off];
        }
    }
}

// ---------------------------------------------------------------------------
// In-place bf16 scan, fp32 carry. One thread per pair of adjacent d-channels
// (packed u32 loads/stores, fully coalesced along d).
// ---------------------------------------------------------------------------
__global__ __launch_bounds__(256)
void lru_scan(const float* __restrict__ lambda_log, unsigned int* __restrict__ gu)
{
    const int i  = blockIdx.x * blockDim.x + threadIdx.x;   // 0 .. B*D/2-1
    const int b  = i >> 9;                                  // / (D/2)
    const int dp = i & 511;
    const float lam0 = __expf(lambda_log[2 * dp]);
    const float lam1 = __expf(lambda_log[2 * dp + 1]);

    unsigned int* p = gu + (size_t)b * L_DIM * (D_DIM / 2) + dp;
    float h0 = 0.f, h1 = 0.f;
#pragma unroll 1
    for (int t = 0; t < L_DIM; t += 4) {
        const unsigned int u0 = p[0 * (D_DIM / 2)];
        const unsigned int u1 = p[1 * (D_DIM / 2)];
        const unsigned int u2 = p[2 * (D_DIM / 2)];
        const unsigned int u3 = p[3 * (D_DIM / 2)];
        h0 = fmaf(lam0, h0, bf2f(u0 & 0xFFFFu)); h1 = fmaf(lam1, h1, bf2f(u0 >> 16));
        p[0 * (D_DIM / 2)] = pack_bf2(h0, h1);
        h0 = fmaf(lam0, h0, bf2f(u1 & 0xFFFFu)); h1 = fmaf(lam1, h1, bf2f(u1 >> 16));
        p[1 * (D_DIM / 2)] = pack_bf2(h0, h1);
        h0 = fmaf(lam0, h0, bf2f(u2 & 0xFFFFu)); h1 = fmaf(lam1, h1, bf2f(u2 >> 16));
        p[2 * (D_DIM / 2)] = pack_bf2(h0, h1);
        h0 = fmaf(lam0, h0, bf2f(u3 & 0xFFFFu)); h1 = fmaf(lam1, h1, bf2f(u3 >> 16));
        p[3 * (D_DIM / 2)] = pack_bf2(h0, h1);
        p += 4 * (D_DIM / 2);
    }
}

extern "C" void kernel_launch(void* const* d_in, const int* in_sizes, int n_in,
                              void* d_out, int out_size, void* d_ws, size_t ws_size,
                              hipStream_t stream) {
    (void)in_sizes; (void)n_in; (void)out_size; (void)ws_size;
    const float* x          = (const float*)d_in[0];
    const float* lambda_log = (const float*)d_in[1];
    const float* gamma      = (const float*)d_in[2];
    const float* W_in       = (const float*)d_in[3];
    const float* b_in       = (const float*)d_in[4];
    const float* W_out      = (const float*)d_in[5];
    const float* b_out      = (const float*)d_in[6];
    float* out = (float*)d_out;

    // workspace layout: gu/y bf16 (128 MB) | W_in bf16 (2 MB) | W_out bf16 (2 MB)
    unsigned short* gu   = (unsigned short*)d_ws;
    unsigned short* Wib  = gu + (size_t)M_TOT * D_DIM;
    unsigned short* Wob  = Wib + (size_t)D_DIM * D_DIM;

    dim3 blk(256);
    dim3 grid_gemm(M_TOT / BM, D_DIM / BN);                 // 512 x 16

    w_to_bf16<<<(D_DIM * D_DIM / 2) / 256, blk, 0, stream>>>(W_in,  (unsigned int*)Wib);
    w_to_bf16<<<(D_DIM * D_DIM / 2) / 256, blk, 0, stream>>>(W_out, (unsigned int*)Wob);
    lru_gemm1<<<grid_gemm, blk, 0, stream>>>(x, Wib, b_in, gamma, gu);
    lru_scan<<<(B_DIM * D_DIM / 2) / 256, blk, 0, stream>>>(lambda_log, (unsigned int*)gu);
    lru_gemm2<<<grid_gemm, blk, 0, stream>>>(gu, Wob, b_out, x, out);
}
